// CrossAttention_70111046140013
// MI455X (gfx1250) — matrix operand order repaired and checked by reading, not compiled, not run on hardware
//
#include <hip/hip_runtime.h>

// ---------------------------------------------------------------------------
// CDNA5 (gfx1250) cross-attention: bf16 WMMA GEMMs + flash-style attention.
// Wave32; WMMA 16x16x32 bf16 -> f32 accumulate. bf16 intermediates; async
// global->LDS staging (ASYNCcnt) where the toolchain exposes the builtins.
// ---------------------------------------------------------------------------

typedef __attribute__((ext_vector_type(16))) __bf16          v16bf;
typedef __attribute__((ext_vector_type(8)))  float           v8f;
typedef __attribute__((ext_vector_type(8)))  unsigned short  v8us;
typedef __attribute__((ext_vector_type(4)))  int             v4i;

union FragBF {
    v16bf bf;
    v8us  us[2];
};

__device__ __forceinline__ unsigned short f2bf(float f) {
    return __builtin_bit_cast(unsigned short, (__bf16)f);
}
__device__ __forceinline__ unsigned int f2bf2(float lo, float hi) {
    return (unsigned int)f2bf(lo) | ((unsigned int)f2bf(hi) << 16);
}
__device__ __forceinline__ float bf2f(unsigned short u) {
    union { unsigned int i; float f; } c; c.i = ((unsigned int)u) << 16;
    return c.f;
}

__device__ __forceinline__ v8f wmma_bf16(const FragBF& a, const FragBF& b, v8f c) {
    return __builtin_amdgcn_wmma_f32_16x16x32_bf16(
        false, a.bf, false, b.bf, (short)0, c, false, false);
}

// ---- gfx1250 async global->LDS (ASYNCcnt) with compile-safe fallback -------
#if __has_builtin(__builtin_amdgcn_global_load_async_to_lds_b128) && \
    __has_builtin(__builtin_amdgcn_s_wait_asynccnt)
#define USE_ASYNC_LDS 1
#else
#define USE_ASYNC_LDS 0
#endif

__device__ __forceinline__ void copy16_to_lds(const unsigned short* g, unsigned short* l) {
#if USE_ASYNC_LDS
    __builtin_amdgcn_global_load_async_to_lds_b128(
        (__attribute__((address_space(1))) v4i*)g,
        (__attribute__((address_space(3))) v4i*)l, 0, 0);
#else
    *reinterpret_cast<uint4*>(l) = *reinterpret_cast<const uint4*>(g);
#endif
}
__device__ __forceinline__ void async_lds_wait() {
#if USE_ASYNC_LDS
    __builtin_amdgcn_s_wait_asynccnt(0);
#endif
}

// ---------------------------------------------------------------------------
// Generic GEMM: C[M,N] = A[M,K] @ W[K,N] (+ bias); f32 in, f32 or bf16 out.
// 128 threads (4 waves). Tile BM=64 x BN=64, BK=32, double-buffered LDS.
// ---------------------------------------------------------------------------
#define BM 64
#define BN 64
#define BK 32
#define LDA 40   // BK + 8 pad (bf16 elems); row stride 80B (16B aligned)
#define LDB 40

__device__ __forceinline__ void store_out(float* p, float v)          { *p = v; }
__device__ __forceinline__ void store_out(unsigned short* p, float v) { *p = f2bf(v); }

__device__ __forceinline__ void gemm_gload(const float* __restrict__ A,
                                           const float* __restrict__ W,
                                           int K, int N, int m0, int n0, int k0,
                                           int t, float4 aReg[4], float4 bReg[4])
{
#pragma unroll
    for (int i = 0; i < 4; ++i) {
        int slot = t + i * 128;
        int arow = slot >> 3, akc = (slot & 7) << 2;
        aReg[i] = *reinterpret_cast<const float4*>(&A[(size_t)(m0 + arow) * K + k0 + akc]);
        int brow = slot >> 4, bnc = (slot & 15) << 2;
        bReg[i] = *reinterpret_cast<const float4*>(&W[(size_t)(k0 + brow) * N + n0 + bnc]);
    }
}

__device__ __forceinline__ void gemm_store_lds(unsigned short* As, unsigned short* Bs,
                                               int t, const float4 aReg[4],
                                               const float4 bReg[4])
{
#pragma unroll
    for (int i = 0; i < 4; ++i) {
        int slot = t + i * 128;
        int arow = slot >> 3, akc = (slot & 7) << 2;
        unsigned int* dst = reinterpret_cast<unsigned int*>(&As[arow * LDA + akc]);
        dst[0] = f2bf2(aReg[i].x, aReg[i].y);
        dst[1] = f2bf2(aReg[i].z, aReg[i].w);
        int brow = slot >> 4, bnc = (slot & 15) << 2;
        Bs[(bnc + 0) * LDB + brow] = f2bf(bReg[i].x);
        Bs[(bnc + 1) * LDB + brow] = f2bf(bReg[i].y);
        Bs[(bnc + 2) * LDB + brow] = f2bf(bReg[i].z);
        Bs[(bnc + 3) * LDB + brow] = f2bf(bReg[i].w);
    }
}

template <bool HAS_BIAS, typename OUT_T>
__global__ __launch_bounds__(128)
void gemm_bf16_wmma_kernel(const float* __restrict__ A,
                           const float* __restrict__ W,
                           const float* __restrict__ bias,
                           OUT_T* __restrict__ C,
                           int M, int K, int N)
{
    __shared__ unsigned short Asub[2][BM * LDA];
    __shared__ unsigned short Bsub[2][BN * LDB];   // [n][k] (transposed)

    const int t    = threadIdx.x;
    const int lane = t & 31;
    const int wv   = t >> 5;
    const int m0   = blockIdx.x * BM;
    const int n0   = blockIdx.y * BN;

    v8f acc[4];
#pragma unroll
    for (int i = 0; i < 4; ++i)
#pragma unroll
        for (int j = 0; j < 8; ++j) acc[i][j] = 0.0f;

    float4 aReg[4], bReg[4];
    gemm_gload(A, W, K, N, m0, n0, 0, t, aReg, bReg);
    gemm_store_lds(&Asub[0][0], &Bsub[0][0], t, aReg, bReg);
    __syncthreads();

    int p = 0;
    for (int k0 = 0; k0 < K; k0 += BK) {
        const bool more = (k0 + BK) < K;
        if (more) gemm_gload(A, W, K, N, m0, n0, k0 + BK, t, aReg, bReg);

        const unsigned short* As = &Asub[p][0];
        const unsigned short* Bs = &Bsub[p][0];
        FragBF a;
        {
            int off = (wv * 16 + (lane & 15)) * LDA + ((lane >> 4) << 3);
            a.us[0] = *reinterpret_cast<const v8us*>(&As[off]);
            a.us[1] = *reinterpret_cast<const v8us*>(&As[off + 16]);
        }
        FragBF b[4];
#pragma unroll
        for (int nt = 0; nt < 4; ++nt) {
            int off = (nt * 16 + (lane & 15)) * LDB + ((lane >> 4) << 3);
            b[nt].us[0] = *reinterpret_cast<const v8us*>(&Bs[off]);
            b[nt].us[1] = *reinterpret_cast<const v8us*>(&Bs[off + 16]);
        }
#pragma unroll
        for (int nt = 0; nt < 4; ++nt)
            acc[nt] = wmma_bf16(a, b[nt], acc[nt]);

        if (more) gemm_store_lds(&Asub[p ^ 1][0], &Bsub[p ^ 1][0], t, aReg, bReg);
        __syncthreads();
        p ^= 1;
    }

#pragma unroll
    for (int nt = 0; nt < 4; ++nt) {
        int col  = n0 + nt * 16 + (lane & 15);
        float bv = HAS_BIAS ? bias[col] : 0.0f;
        int rbase = m0 + wv * 16 + ((lane >> 4) << 3);
#pragma unroll
        for (int r = 0; r < 8; ++r)
            store_out(&C[(size_t)(rbase + r) * N + col], acc[nt][r] + bv);
    }
}

// ---------------------------------------------------------------------------
// RoPE in place on bf16 data (rows x rowStride); optional output scale
// (0.125 = 1/sqrt(64) folded into Q exactly).
// ---------------------------------------------------------------------------
__global__ void rope_bf16_kernel(unsigned short* __restrict__ data,
                                 const float* __restrict__ thetas,
                                 int rows, int rowStride, int Npos, float scale)
{
    int idx = blockIdx.x * blockDim.x + threadIdx.x;
    if (idx >= rows * 512) return;         // rows * 16 heads * 32 pairs
    int row = idx >> 9;
    int rem = idx & 511;
    int hh  = rem >> 5;
    int j   = rem & 31;
    int pos = row & (Npos - 1);
    size_t base = (size_t)row * rowStride + hh * 64;
    float x1 = bf2f(data[base + j]);
    float x2 = bf2f(data[base + j + 32]);
    float t1 = thetas[pos * 64 + j];
    float t2 = thetas[pos * 64 + j + 32];
    data[base + j]      = f2bf((x1 * cosf(t1) - x2 * sinf(t1)) * scale);
    data[base + j + 32] = f2bf((x2 * cosf(t2) + x1 * sinf(t2)) * scale);
}

// ---------------------------------------------------------------------------
// Flash attention: one WG = (b, h, 64-query tile). 128 threads / 4 waves.
// Q already scaled; all operands bf16. Q/K tiles staged with async
// global->LDS; V staged via registers into transposed LDS layout.
// ---------------------------------------------------------------------------
#define QLD 72   // 64 + 8 pad (bf16 elems); row stride 144B (16B aligned)

__global__ __launch_bounds__(128)
void attn_kernel(const unsigned short* __restrict__ Q,   // (B*Nx, D) bf16
                 const unsigned short* __restrict__ KV,  // (B*Nc, 2D) bf16
                 float* __restrict__ O,                  // (B*Nx, D) f32
                 int Nx, int Nc, int D)
{
    __shared__ unsigned short Qs[64 * QLD];            // [q][d]
    __shared__ unsigned short Ks[64 * QLD];            // [key][d]
    __shared__ unsigned short Vt[64 * QLD];            // [d][key] (transposed)
    __shared__ unsigned short Ps[4 * 16 * QLD];        // per-wave P tile [row][key]

    const int t    = threadIdx.x;
    const int lane = t & 31;
    const int wv   = t >> 5;
    const int bid  = blockIdx.x;
    const int qt   = bid & 15;            // Nx/64 = 16 query tiles
    const int h    = (bid >> 4) & 15;     // 16 heads
    const int b    = bid >> 8;
    const int q0   = qt * 64;
    const size_t strideKV = 2 * (size_t)D;

    // ---- stage Q tile: straight bf16 copy, async global->LDS
#pragma unroll
    for (int i = 0; i < 4; ++i) {
        int slot = t + i * 128;
        int row  = slot >> 3;
        int ch   = (slot & 7) << 3;     // 8 bf16 = 16B chunks
        copy16_to_lds(&Q[((size_t)b * Nx + q0 + row) * D + h * 64 + ch],
                      &Qs[row * QLD + ch]);
    }
    async_lds_wait();
    __syncthreads();

    FragBF aq[2];
#pragma unroll
    for (int c = 0; c < 2; ++c) {
        int off = (wv * 16 + (lane & 15)) * QLD + c * 32 + ((lane >> 4) << 3);
        aq[c].us[0] = *reinterpret_cast<const v8us*>(&Qs[off]);
        aq[c].us[1] = *reinterpret_cast<const v8us*>(&Qs[off + 16]);
    }

    float mrow[8], lrow[8];
    v8f acc[4];
#pragma unroll
    for (int r = 0; r < 8; ++r) { mrow[r] = -3.0e38f; lrow[r] = 0.0f; }
#pragma unroll
    for (int i = 0; i < 4; ++i)
#pragma unroll
        for (int j = 0; j < 8; ++j) acc[i][j] = 0.0f;

    const int nTiles = Nc >> 6;
    for (int kt = 0; kt < nTiles; ++kt) {
        const size_t rowBase = (size_t)b * Nc + (size_t)kt * 64;

        // ---- V tile into registers BEFORE the barrier (latency hiding)
        uint4 vReg[4];
#pragma unroll
        for (int i = 0; i < 4; ++i) {
            int slot = t + i * 128;
            int row  = slot >> 3;
            int ch   = (slot & 7) << 3;
            vReg[i] = *reinterpret_cast<const uint4*>(
                &KV[(rowBase + row) * strideKV + D + h * 64 + ch]);
        }

        __syncthreads();   // previous tile's readers done; LDS reusable

        // ---- K tile: async global->LDS (no VGPR round trip)
#pragma unroll
        for (int i = 0; i < 4; ++i) {
            int slot = t + i * 128;
            int row  = slot >> 3;
            int ch   = (slot & 7) << 3;
            copy16_to_lds(&KV[(rowBase + row) * strideKV + h * 64 + ch],
                          &Ks[row * QLD + ch]);
        }
        // ---- V tile: transposed scatter into LDS (bf16, no conversion)
#pragma unroll
        for (int i = 0; i < 4; ++i) {
            int slot = t + i * 128;
            int row  = slot >> 3;
            int d0   = (slot & 7) << 3;
            const unsigned short* pv = reinterpret_cast<const unsigned short*>(&vReg[i]);
#pragma unroll
            for (int j = 0; j < 8; ++j)
                Vt[(d0 + j) * QLD + row] = pv[j];
        }
        async_lds_wait();
        __syncthreads();

        // ---- S = Q @ K^T : preload all 8 B-fragments, then 8 WMMAs
        FragBF bk[4][2];
#pragma unroll
        for (int nt = 0; nt < 4; ++nt)
#pragma unroll
            for (int c = 0; c < 2; ++c) {
                int off = (nt * 16 + (lane & 15)) * QLD + c * 32 + ((lane >> 4) << 3);
                bk[nt][c].us[0] = *reinterpret_cast<const v8us*>(&Ks[off]);
                bk[nt][c].us[1] = *reinterpret_cast<const v8us*>(&Ks[off + 16]);
            }
        v8f S[4];
#pragma unroll
        for (int i = 0; i < 4; ++i)
#pragma unroll
            for (int j = 0; j < 8; ++j) S[i][j] = 0.0f;
#pragma unroll
        for (int c = 0; c < 2; ++c)
#pragma unroll
            for (int nt = 0; nt < 4; ++nt)
                S[nt] = wmma_bf16(aq[c], bk[nt][c], S[nt]);

        // ---- online softmax (rows M = r + 8*(lane>>4); 16-lane butterflies)
        float mx[8];
#pragma unroll
        for (int r = 0; r < 8; ++r) {
            float v = fmaxf(fmaxf(S[0][r], S[1][r]), fmaxf(S[2][r], S[3][r]));
#pragma unroll
            for (int off = 8; off >= 1; off >>= 1)
                v = fmaxf(v, __shfl_xor(v, off, 32));
            mx[r] = v;
        }
        float corr[8];
#pragma unroll
        for (int r = 0; r < 8; ++r) {
            float mn = fmaxf(mrow[r], mx[r]);
            corr[r]  = __expf(mrow[r] - mn);
            mrow[r]  = mn;
        }
#pragma unroll
        for (int nt = 0; nt < 4; ++nt)
#pragma unroll
            for (int r = 0; r < 8; ++r) acc[nt][r] *= corr[r];

        float rs[8];
#pragma unroll
        for (int r = 0; r < 8; ++r) rs[r] = 0.0f;
        const int prow0 = ((lane >> 4) << 3);
#pragma unroll
        for (int nt = 0; nt < 4; ++nt)
#pragma unroll
            for (int r = 0; r < 8; ++r) {
                float pv = __expf(S[nt][r] - mrow[r]);
                rs[r] += pv;
                Ps[wv * (16 * QLD) + (prow0 + r) * QLD + nt * 16 + (lane & 15)] = f2bf(pv);
            }
#pragma unroll
        for (int r = 0; r < 8; ++r) {
            float v = rs[r];
#pragma unroll
            for (int off = 8; off >= 1; off >>= 1)
                v += __shfl_xor(v, off, 32);
            lrow[r] = lrow[r] * corr[r] + v;
        }

        // ---- O += P @ V : preload 2 A-frags + 8 B-frags, then 8 WMMAs
        FragBF ap[2], bv[4][2];
#pragma unroll
        for (int c = 0; c < 2; ++c) {
            int offp = wv * (16 * QLD) + (lane & 15) * QLD + c * 32 + ((lane >> 4) << 3);
            ap[c].us[0] = *reinterpret_cast<const v8us*>(&Ps[offp]);
            ap[c].us[1] = *reinterpret_cast<const v8us*>(&Ps[offp + 16]);
        }
#pragma unroll
        for (int nt = 0; nt < 4; ++nt)
#pragma unroll
            for (int c = 0; c < 2; ++c) {
                int offv = (nt * 16 + (lane & 15)) * QLD + c * 32 + ((lane >> 4) << 3);
                bv[nt][c].us[0] = *reinterpret_cast<const v8us*>(&Vt[offv]);
                bv[nt][c].us[1] = *reinterpret_cast<const v8us*>(&Vt[offv + 16]);
            }
#pragma unroll
        for (int c = 0; c < 2; ++c)
#pragma unroll
            for (int nt = 0; nt < 4; ++nt)
                acc[nt] = wmma_bf16(ap[c], bv[nt][c], acc[nt]);
    }

    // ---- finalize and store f32 into (B*Nx, D) for the projection GEMM
    float inv[8];
#pragma unroll
    for (int r = 0; r < 8; ++r) inv[r] = 1.0f / lrow[r];
#pragma unroll
    for (int nt = 0; nt < 4; ++nt) {
        int col = h * 64 + nt * 16 + (lane & 15);
        size_t rbase = (size_t)b * Nx + q0 + wv * 16 + ((lane >> 4) << 3);
#pragma unroll
        for (int r = 0; r < 8; ++r)
            O[(rbase + r) * D + col] = acc[nt][r] * inv[r];
    }
}

// ---------------------------------------------------------------------------
// Host launcher
// ---------------------------------------------------------------------------
extern "C" void kernel_launch(void* const* d_in, const int* in_sizes, int n_in,
                              void* d_out, int out_size, void* d_ws, size_t ws_size,
                              hipStream_t stream)
{
    (void)in_sizes; (void)n_in; (void)out_size; (void)ws_size;

    const float* x    = (const float*)d_in[0];   // (4,1024,1024)
    const float* c    = (const float*)d_in[1];   // (4,2048,1024)
    const float* xpe  = (const float*)d_in[2];   // (1024,64)
    const float* cpe  = (const float*)d_in[3];   // (2048,64)
    const float* Wq   = (const float*)d_in[4];   // (1024,1024)
    const float* Wkv  = (const float*)d_in[5];   // (1024,2048)
    const float* Wpr  = (const float*)d_in[6];   // (1024,1024)
    const float* bpr  = (const float*)d_in[7];   // (1024,)
    float* out = (float*)d_out;                  // (4,1024,1024)

    const int B = 4, Nx = 1024, Nc = 2048, D = 1024, H = 16;

    unsigned short* Qw  = (unsigned short*)d_ws;          // (B*Nx, D)  bf16  8 MB
    unsigned short* KVw = Qw + (size_t)B * Nx * D;        // (B*Nc, 2D) bf16 32 MB
    float*          Aw  = (float*)(KVw + (size_t)B * Nc * 2 * D);  // f32 16 MB

    dim3 blk(128);

    // Q = x @ Wq (bf16 out)
    gemm_bf16_wmma_kernel<false, unsigned short>
        <<<dim3(B * Nx / 64, D / 64), blk, 0, stream>>>(x, Wq, nullptr, Qw, B * Nx, D, D);
    // KV = c @ Wkv (bf16 out)
    gemm_bf16_wmma_kernel<false, unsigned short>
        <<<dim3(B * Nc / 64, 2 * D / 64), blk, 0, stream>>>(c, Wkv, nullptr, KVw, B * Nc, D, 2 * D);

    // RoPE in place; Q also picks up the exact 1/sqrt(hd) = 0.125 scale
    rope_bf16_kernel<<<(B * Nx * 512 + 255) / 256, 256, 0, stream>>>(
        Qw, xpe, B * Nx, D, Nx, 0.125f);
    rope_bf16_kernel<<<(B * Nc * 512 + 255) / 256, 256, 0, stream>>>(
        KVw, cpe, B * Nc, 2 * D, Nc, 1.0f);

    // Flash attention -> (B*Nx, D) f32
    attn_kernel<<<B * H * (Nx / 64), blk, 0, stream>>>(Qw, KVw, Aw, Nx, Nc, D);

    // out = Aw @ Wproj + bproj (f32 out)
    gemm_bf16_wmma_kernel<true, float>
        <<<dim3(B * Nx / 64, D / 64), blk, 0, stream>>>(Aw, Wpr, bpr, out, B * Nx, D, D);
}
